// SequenceClassification_69965017252743
// MI455X (gfx1250) — matrix operand order repaired and checked by reading, not compiled, not run on hardware
//
#include <hip/hip_runtime.h>

// ---------------------------------------------------------------------------
// CDNA5 (gfx1250) implementation of the BiDAF-style sequence classifier.
// All dense GEMMs run on v_wmma_f32_16x16x32_bf16 (bf16 operands, f32 accum).
// Inner GEMM loop is branch-free; OOB fragment rows are clamped (their WMMA
// results map only to discarded output rows/cols). All global accesses use
// explicit address_space(1) pointers with byte offsets so loads lower to
// global_load_b128 with saddr + 32-bit voffset + immediate — no flat ops and
// no per-load 64-bit address arithmetic in the K loop.
// ---------------------------------------------------------------------------

typedef __attribute__((ext_vector_type(16))) __bf16 v16bf;
typedef __attribute__((ext_vector_type(8)))  float  v8f;
typedef __attribute__((ext_vector_type(4)))  unsigned int u32x4;  // builtin, AS-agnostic

typedef const char           __attribute__((address_space(1)))* gbytep;
typedef const u32x4          __attribute__((address_space(1)))* gvecp;
typedef float                __attribute__((address_space(1)))* gfp;
typedef unsigned short       __attribute__((address_space(1)))* gu16p;
typedef const float          __attribute__((address_space(1)))* gcfp;

#define DD   256
#define LL   256
#define BB   16
#define HH   8
#define DH   32
#define NTOK (BB * LL)

__device__ __forceinline__ unsigned short f2bf(float f) {
  union { float f; unsigned u; } x; x.f = f;
  unsigned r = x.u + 0x7FFFu + ((x.u >> 16) & 1u);   // round-to-nearest-even
  return (unsigned short)(r >> 16);
}

union Frag { v16bf v; u32x4 u[2]; };

// ---------------------------------------------------------------------------
// Generic batched GEMM:  C[z] = act( scale * A[z] @ W[z]^T + bias )
//   A: [M,K] bf16 row-major (lda elems), W: [N,K] bf16 row-major (ldw elems)
//   Cf (f32) / Cb (bf16) written if non-null, row stride ldc.
// Block = 128 threads = 4 waves. Wave tile 32(M) x 64(N); block tile 32 x 256.
// Requires M%16==0, K%32==0; N arbitrary (OOB rows clamped, stores guarded).
// ---------------------------------------------------------------------------
__global__ __launch_bounds__(128) void k_gemm_bf16(
    float* __restrict__ Cf, unsigned short* __restrict__ Cb,
    const unsigned short* __restrict__ A, const unsigned short* __restrict__ W,
    const float* __restrict__ bias,
    int M, int N, int K, int lda, int ldw, int ldc,
    long long sAz, long long sWz, long long sCz, float scale, int relu)
{
  const int z = blockIdx.z;
  gbytep Ab = (gbytep)(A + (size_t)z * sAz);
  gbytep Wb = (gbytep)(W + (size_t)z * sWz);
  const size_t coff = (size_t)z * sCz;

  const int lane  = threadIdx.x & 31;
  const int wave  = threadIdx.x >> 5;
  const int half  = lane >> 4;
  const int lr    = lane & 15;
  const int tileM = blockIdx.y * 32;
  const int nbase = blockIdx.x * 256 + wave * 64;

  v8f acc[8];
#pragma unroll
  for (int i = 0; i < 8; ++i)
#pragma unroll
    for (int e = 0; e < 8; ++e) acc[i][e] = 0.0f;

  const bool m2 = (tileM + 16) < M;       // second M-frag stored (M%16==0)
  const int  m0 = tileM + lr;
  const int  m1 = m0 + 16;
  const int  mc1 = (m1 < M) ? m1 : (M - 1);   // clamp: results discarded

  // Byte offsets (scalar base + 32-bit voffset + imm addressing).
  // A frag: lane holds row m, k = {half*8 .. +7} U {16+half*8 .. +7}
  // B frag: lane holds W row n (= output col), k = {half*8 .. +7} U {16+half*8 .. +7}
  const unsigned oa0 = ((unsigned)m0  * (unsigned)lda + (unsigned)(half * 8)) * 2u;
  const unsigned oa1 = ((unsigned)mc1 * (unsigned)lda + (unsigned)(half * 8)) * 2u;
  unsigned ob[4];
#pragma unroll
  for (int j = 0; j < 4; ++j) {
    int n  = nbase + j * 16 + lr;
    int nc = (n < N) ? n : (N - 1);           // clamp: results discarded
    ob[j] = ((unsigned)nc * (unsigned)ldw + (unsigned)(half * 8)) * 2u;
  }

  const unsigned kend = (unsigned)K * 2u;      // bytes; K step 32 elems = 64 B
  for (unsigned kb = 0; kb < kend; kb += 64u) {
    Frag a0, a1, b[4];
    a0.u[0] = *(gvecp)(Ab + (oa0 + kb));
    a0.u[1] = *(gvecp)(Ab + (oa0 + kb + 32u));
    a1.u[0] = *(gvecp)(Ab + (oa1 + kb));
    a1.u[1] = *(gvecp)(Ab + (oa1 + kb + 32u));
#pragma unroll
    for (int j = 0; j < 4; ++j) {
      b[j].u[0] = *(gvecp)(Wb + (ob[j] + kb));
      b[j].u[1] = *(gvecp)(Wb + (ob[j] + kb + 32u));
    }
#pragma unroll
    for (int j = 0; j < 4; ++j) {
      acc[j]     = __builtin_amdgcn_wmma_f32_16x16x32_bf16(
                     false, a0.v, false, b[j].v, (short)0, acc[j],     false, false);
      acc[4 + j] = __builtin_amdgcn_wmma_f32_16x16x32_bf16(
                     false, a1.v, false, b[j].v, (short)0, acc[4 + j], false, false);
    }
  }

  // Epilogue. C/D layout: lane -> n = lr, vgpr v -> m = v + 8*half.
  gfp   Cfg   = (gfp)Cf;
  gu16p Cbg   = (gu16p)Cb;
  gcfp  biasg = (gcfp)bias;
#pragma unroll
  for (int j = 0; j < 4; ++j) {
    const int n = nbase + j * 16 + lr;
    if (n >= N) continue;
    const float bv = bias ? biasg[n] : 0.0f;
#pragma unroll
    for (int v = 0; v < 8; ++v) {
      const int m = tileM + v + 8 * half;
      float x = acc[j][v] * scale + bv;
      if (relu) x = fmaxf(x, 0.0f);
      size_t o = coff + (size_t)m * ldc + n;
      if (Cf) Cfg[o] = x;
      if (Cb) Cbg[o] = f2bf(x);
      if (m2) {
        float y = acc[4 + j][v] * scale + bv;
        if (relu) y = fmaxf(y, 0.0f);
        size_t o2 = coff + (size_t)(m + 16) * ldc + n;
        if (Cf) Cfg[o2] = y;
        if (Cb) Cbg[o2] = f2bf(y);
      }
    }
  }
}

// --------------------------- elementwise / reductions ----------------------

__global__ __launch_bounds__(256) void k_cvt(unsigned short* __restrict__ o,
                                             const float* __restrict__ x, long long n) {
  long long g = (long long)blockIdx.x * 256 + threadIdx.x;
  if (g < n) o[g] = f2bf(x[g]);
}

__global__ __launch_bounds__(256) void k_gather(float* of, unsigned short* ob,
                                                const float* emb, const int* idx) {
  int t = blockIdx.x, d = threadIdx.x;
  float v = emb[(size_t)idx[t] * DD + d];
  of[(size_t)t * DD + d] = v;
  ob[(size_t)t * DD + d] = f2bf(v);
}

__global__ __launch_bounds__(256) void k_split_qkv(unsigned short* q, unsigned short* k,
                                                   unsigned short* vt, const float* qkv) {
  int g = blockIdx.x * 256 + threadIdx.x;          // over B*H*L*dh = 1<<20
  int d = g & 31, t = (g >> 5) & 255, h = (g >> 13) & 7, b = g >> 16;
  size_t row = (size_t)(b * LL + t) * (3 * DD) + h * DH + d;
  int zh = b * HH + h;
  q [(size_t)(zh * LL + t) * DH + d] = f2bf(qkv[row]);
  k [(size_t)(zh * LL + t) * DH + d] = f2bf(qkv[row + DD]);
  vt[(size_t)(zh * DH + d) * LL + t] = f2bf(qkv[row + 2 * DD]);   // V^T per head
}

__global__ __launch_bounds__(256) void k_merge(unsigned short* om, const float* oh) {
  int g = blockIdx.x * 256 + threadIdx.x;          // over B*H*L*dh
  int d = g & 31, t = (g >> 5) & 255, h = (g >> 13) & 7, b = g >> 16;
  om[(size_t)(b * LL + t) * DD + h * DH + d] =
      f2bf(oh[(size_t)((b * HH + h) * LL + t) * DH + d]);
}

__global__ __launch_bounds__(256) void k_add_ln(float* of, unsigned short* ob,
                                                const float* x, const float* res,
                                                const float* g, const float* b) {
  __shared__ float red[256];
  int row = blockIdx.x, d = threadIdx.x;
  size_t o = (size_t)row * DD + d;
  float v = x[o] + (res ? res[o] : 0.0f);
  red[d] = v; __syncthreads();
  for (int s = 128; s > 0; s >>= 1) { if (d < s) red[d] += red[d + s]; __syncthreads(); }
  float mean = red[0] * (1.0f / DD); __syncthreads();
  float c = v - mean;
  red[d] = c * c; __syncthreads();
  for (int s = 128; s > 0; s >>= 1) { if (d < s) red[d] += red[d + s]; __syncthreads(); }
  float y = c * rsqrtf(red[0] * (1.0f / DD) + 1e-5f) * g[d] + b[d];
  of[o] = y;
  if (ob) ob[o] = f2bf(y);
}

__global__ __launch_bounds__(256) void k_softmax(float* of, unsigned short* ob,
                                                 const float* in) {
  __shared__ float red[256];
  int row = blockIdx.x, d = threadIdx.x;
  size_t o = (size_t)row * 256 + d;
  float v = in[o];
  red[d] = v; __syncthreads();
  for (int s = 128; s > 0; s >>= 1) { if (d < s) red[d] = fmaxf(red[d], red[d + s]); __syncthreads(); }
  float mx = red[0]; __syncthreads();
  float e = __expf(v - mx);
  red[d] = e; __syncthreads();
  for (int s = 128; s > 0; s >>= 1) { if (d < s) red[d] += red[d + s]; __syncthreads(); }
  float p = e / red[0];
  if (of) of[o] = p;
  if (ob) ob[o] = f2bf(p);
}

__global__ __launch_bounds__(256) void k_argmax(float* out, const float* in) {
  __shared__ float rv[256]; __shared__ int ri[256];
  int row = blockIdx.x, d = threadIdx.x;
  rv[d] = in[(size_t)row * 256 + d]; ri[d] = d; __syncthreads();
  for (int s = 128; s > 0; s >>= 1) {
    if (d < s) {
      float ov = rv[d + s]; int oi = ri[d + s];
      if (ov > rv[d] || (ov == rv[d] && oi < ri[d])) { rv[d] = ov; ri[d] = oi; }
    }
    __syncthreads();
  }
  if (d == 0) out[row] = (float)ri[0];
}

// q_res/ctx build: [enc | emb] -> f32 + bf16 (+ transposed bf16 ctx^T, + rowsum)
__global__ __launch_bounds__(256) void k_concat(float* rf, unsigned short* rb,
                                                unsigned short* ctxT, float* rowsum,
                                                const float* enc, const float* embx) {
  __shared__ float red[256];
  int n = blockIdx.x, d = threadIdx.x;
  float v0 = enc [(size_t)n * DD + d];
  float v1 = embx[(size_t)n * DD + d];
  size_t base = (size_t)n * 512;
  rf[base + d] = v0;        rf[base + 256 + d] = v1;
  rb[base + d] = f2bf(v0);  rb[base + 256 + d] = f2bf(v1);
  if (ctxT) {
    int b = n >> 8, t = n & 255;
    size_t tb = (size_t)b * 512 * LL;
    ctxT[tb + (size_t)d * LL + t]         = f2bf(v0);
    ctxT[tb + (size_t)(256 + d) * LL + t] = f2bf(v1);
  }
  red[d] = v0 + v1; __syncthreads();
  for (int s = 128; s > 0; s >>= 1) { if (d < s) red[d] += red[d + s]; __syncthreads(); }
  if (d == 0) rowsum[n] = red[0];
}

// trilinear rank-1 terms: S[b,t,j] += sum_d ctx[b,t,:] + sum_d q_res[b,j,:]
__global__ __launch_bounds__(256) void k_sepi(float* S, const float* rs_ctx,
                                              const float* rs_q) {
  int row = blockIdx.x, j = threadIdx.x;   // row = b*L + t
  int b = row >> 8;
  S[(size_t)row * 256 + j] += rs_ctx[row] + rs_q[b * 256 + j];
}

__global__ __launch_bounds__(256) void k_q2c(float* q2c, const float* beta,
                                             const float* qres) {
  int g = blockIdx.x * 256 + threadIdx.x;  // over B*512
  int d = g & 511, b = g >> 9;
  float s = 0.0f;
  for (int t = 0; t < LL; ++t)
    s += beta[b * LL + t] * qres[(size_t)(b * LL + t) * 512 + d];
  q2c[g] = s;
}

// cat = [q_res | ctx | c2q | ctx*c2q | ctx*q2c | ctx]  (12D = 3072), bf16
__global__ __launch_bounds__(256) void k_cat(unsigned short* cat, const float* qres,
                                             const float* ctx, const float* c2q,
                                             const float* q2c) {
  size_t g = (size_t)blockIdx.x * 256 + threadIdx.x;  // N*3072
  int d = (int)(g % 3072); size_t n = g / 3072; int b = (int)(n >> 8);
  size_t r = n * 512;
  float v;
  if      (d <  512) v = qres[r + d];
  else if (d < 1024) v = ctx[r + d - 512];
  else if (d < 1536) v = c2q[r + d - 1024];
  else if (d < 2048) v = ctx[r + d - 1536] * c2q[r + d - 1536];
  else if (d < 2560) v = ctx[r + d - 2048] * q2c[(size_t)b * 512 + (d - 2048)];
  else               v = ctx[r + d - 2560];
  cat[g] = f2bf(v);
}

__global__ __launch_bounds__(256) void k_pool(float* rf, unsigned short* rb,
                                              const float* res) {
  int g = blockIdx.x * 256 + threadIdx.x;  // over B*6144
  int c = g % 6144, b = g / 6144;
  float s = 0.0f, mx = -3.4e38f;
  const float* p = res + (size_t)b * LL * 6144 + c;
  for (int t = 0; t < LL; ++t) { float v = p[(size_t)t * 6144]; s += v; mx = fmaxf(mx, v); }
  float mean = s * (1.0f / LL);
  rf[(size_t)b * 12288 + c]        = mean;  rf[(size_t)b * 12288 + 6144 + c] = mx;
  rb[(size_t)b * 12288 + c] = f2bf(mean);   rb[(size_t)b * 12288 + 6144 + c] = f2bf(mx);
}

// ---------------------------------------------------------------------------

extern "C" void kernel_launch(void* const* d_in, const int* in_sizes, int n_in,
                              void* d_out, int out_size, void* d_ws, size_t ws_size,
                              hipStream_t stream) {
  (void)in_sizes; (void)n_in; (void)out_size; (void)ws_size;
  typedef unsigned short u16;

  // ---- inputs ----
  const int*   q_vec = (const int*)d_in[0];
  const int*   d_vec = (const int*)d_in[1];
  const int*   sd_vec= (const int*)d_in[2];
  const float* emb   = (const float*)d_in[3];
  struct Tx { const float *in_w,*in_b,*out_w,*out_b,*w1,*bf1,*w2,*bf2,*g1,*b1,*g2,*b2,*gn,*bn; };
  auto get_tx = [&](int base) {
    Tx t;
    t.in_w=(const float*)d_in[base+0];  t.in_b=(const float*)d_in[base+1];
    t.out_w=(const float*)d_in[base+2]; t.out_b=(const float*)d_in[base+3];
    t.w1=(const float*)d_in[base+4];    t.bf1=(const float*)d_in[base+5];
    t.w2=(const float*)d_in[base+6];    t.bf2=(const float*)d_in[base+7];
    t.g1=(const float*)d_in[base+8];    t.b1=(const float*)d_in[base+9];
    t.g2=(const float*)d_in[base+10];   t.b2=(const float*)d_in[base+11];
    t.gn=(const float*)d_in[base+12];   t.bn=(const float*)d_in[base+13];
    return t;
  };
  Tx tx[3] = { get_tx(4), get_tx(18), get_tx(32) };
  const float* qd_w  = (const float*)d_in[46]; const float* qd_b  = (const float*)d_in[47];
  const float* qsd_w = (const float*)d_in[48]; const float* qsd_b = (const float*)d_in[49];
  const float* res_w = (const float*)d_in[50]; const float* p_res_b=(const float*)d_in[51];
  const float* c_w1  = (const float*)d_in[52]; const float* c_b1  = (const float*)d_in[53];
  const float* c_w2  = (const float*)d_in[54]; const float* c_b2  = (const float*)d_in[55];
  const float* c_w3  = (const float*)d_in[56]; const float* c_b3  = (const float*)d_in[57];
  const float* c_w4  = (const float*)d_in[58]; const float* c_b4  = (const float*)d_in[59];

  // ---- scratch arena ----
  char* ap = (char*)d_ws;
  auto alloc = [&](size_t bytes) -> void* {
    void* r = (void*)ap; ap += (bytes + 255) & ~(size_t)255; return r;
  };
  const size_t N = NTOK;   // 4096 tokens per stream

  // persistent activations
  float* xemb_f[3]; u16* xemb_b[3]; float* resf[3]; u16* resb[3];
  u16* ctxT[2]; float* rowsum[3];
  for (int s = 0; s < 3; ++s) {
    xemb_f[s] = (float*)alloc(N * DD * 4);
    xemb_b[s] = (u16*)  alloc(N * DD * 2);
    resf[s]   = (float*)alloc(N * 512 * 4);
    resb[s]   = (u16*)  alloc(N * 512 * 2);
    rowsum[s] = (float*)alloc(N * 4);
  }
  ctxT[0] = (u16*)alloc(N * 512 * 2);
  ctxT[1] = (u16*)alloc(N * 512 * 2);

  // bf16 weights
  u16 *w_in[3], *w_out[3], *w_f1[3], *w_f2[3];
  for (int s = 0; s < 3; ++s) {
    w_in[s]  = (u16*)alloc((size_t)3*DD*DD*2);
    w_out[s] = (u16*)alloc((size_t)DD*DD*2);
    w_f1[s]  = (u16*)alloc((size_t)2048*DD*2);
    w_f2[s]  = (u16*)alloc((size_t)DD*2048*2);
  }
  u16* w_qd  = (u16*)alloc((size_t)3072*3072*2);
  u16* w_qsd = (u16*)alloc((size_t)3072*3072*2);
  u16* w_res = (u16*)alloc((size_t)6144*6144*2);
  u16* w_c1  = (u16*)alloc((size_t)8192*12288*2);
  u16* w_c2  = (u16*)alloc((size_t)2048*8192*2);
  u16* w_c3  = (u16*)alloc((size_t)1024*2048*2);
  u16* w_c4  = (u16*)alloc((size_t)5*1024*2);

  // encoder scratch (reused per stream)
  float* qkv_f = (float*)alloc(N * 768 * 4);
  u16* qh = (u16*)alloc(N * DD * 2);
  u16* kh = (u16*)alloc(N * DD * 2);
  u16* vT = (u16*)alloc(N * DD * 2);
  float* Sa_f = (float*)alloc((size_t)BB*HH*LL*LL*4);
  u16*   Pa_b = (u16*)  alloc((size_t)BB*HH*LL*LL*2);
  float* Oh_f = (float*)alloc(N * DD * 4);
  u16*   Om_b = (u16*)  alloc(N * DD * 2);
  float* y_f  = (float*)alloc(N * DD * 4);
  float* x1_f = (float*)alloc(N * DD * 4);
  u16*   x1_b = (u16*)  alloc(N * DD * 2);
  float* hf_f = (float*)alloc(N * 2048 * 4);
  u16*   hf_b = (u16*)  alloc(N * 2048 * 2);
  float* f_f  = (float*)alloc(N * DD * 4);
  float* x2_f = (float*)alloc(N * DD * 4);
  float* enc_f= (float*)alloc(N * DD * 4);

  // branch scratch (reused)
  float* Sb_f = (float*)alloc((size_t)BB*LL*LL*4);
  u16*   Pb_b = (u16*)  alloc((size_t)BB*LL*LL*2);
  float* idxf = (float*)alloc(BB * LL * 4);
  float* beta = (float*)alloc(BB * LL * 4);
  float* q2c  = (float*)alloc(BB * 512 * 4);
  float* c2q_f= (float*)alloc(N * 512 * 4);
  u16*   cat_b= (u16*)  alloc(N * 3072 * 2);

  // tail
  u16*   allc_b = (u16*)  alloc(N * 6144 * 2);
  float* rout_f = (float*)alloc(N * 6144 * 4);
  float* r_f    = (float*)alloc((size_t)BB * 12288 * 4);
  u16*   r_b    = (u16*)  alloc((size_t)BB * 12288 * 2);
  u16*   h1_b   = (u16*)  alloc((size_t)BB * 8192 * 2);
  u16*   h2_b   = (u16*)  alloc((size_t)BB * 2048 * 2);
  u16*   h3_b   = (u16*)  alloc((size_t)BB * 1024 * 2);

  // ---- helpers ----
  auto cvt = [&](u16* dst, const float* src, long long n) {
    k_cvt<<<dim3((unsigned)((n + 255) / 256)), dim3(256), 0, stream>>>(dst, src, n);
  };
  auto gemm = [&](float* Cf, u16* Cb, const u16* A, const u16* W, const float* bias,
                  int M, int Nn, int K, int lda, int ldw, int ldc, int batch,
                  long long sA, long long sW, long long sC, float scale, int relu) {
    dim3 g((Nn + 255) / 256, (M + 31) / 32, batch);
    k_gemm_bf16<<<g, dim3(128), 0, stream>>>(Cf, Cb, A, W, bias, M, Nn, K,
                                             lda, ldw, ldc, sA, sW, sC, scale, relu);
  };

  // ---- weight conversion (every call: deterministic) ----
  for (int s = 0; s < 3; ++s) {
    cvt(w_in[s],  tx[s].in_w,  (long long)3*DD*DD);
    cvt(w_out[s], tx[s].out_w, (long long)DD*DD);
    cvt(w_f1[s],  tx[s].w1,    (long long)2048*DD);
    cvt(w_f2[s],  tx[s].w2,    (long long)DD*2048);
  }
  cvt(w_qd,  qd_w,  (long long)3072*3072);
  cvt(w_qsd, qsd_w, (long long)3072*3072);
  cvt(w_res, res_w, (long long)6144*6144);
  cvt(w_c1,  c_w1,  (long long)8192*12288);
  cvt(w_c2,  c_w2,  (long long)2048*8192);
  cvt(w_c3,  c_w3,  (long long)1024*2048);
  cvt(w_c4,  c_w4,  (long long)5*1024);

  // ---- embeddings ----
  const int* vecs[3] = { q_vec, d_vec, sd_vec };
  for (int s = 0; s < 3; ++s)
    k_gather<<<dim3(NTOK), dim3(256), 0, stream>>>(xemb_f[s], xemb_b[s], emb, vecs[s]);

  // ---- encoders + residual concat ----
  const float att_scale = 0.17677669529663687f;  // dh^-0.5
  for (int s = 0; s < 3; ++s) {
    const Tx& T = tx[s];
    // qkv = x @ in_w^T + in_b
    gemm(qkv_f, nullptr, xemb_b[s], w_in[s], T.in_b,
         NTOK, 3*DD, DD, DD, DD, 3*DD, 1, 0, 0, 0, 1.0f, 0);
    k_split_qkv<<<dim3((1 << 20) / 256), dim3(256), 0, stream>>>(qh, kh, vT, qkv_f);
    // S = q k^T * scale   (batched over B*H)
    gemm(Sa_f, nullptr, qh, kh, nullptr, LL, LL, DH, DH, DH, LL,
         BB*HH, (long long)LL*DH, (long long)LL*DH, (long long)LL*LL, att_scale, 0);
    k_softmax<<<dim3(BB*HH*LL), dim3(256), 0, stream>>>(nullptr, Pa_b, Sa_f);
    // O = P V    (W = V^T stored [dh, L])
    gemm(Oh_f, nullptr, Pa_b, vT, nullptr, LL, DH, LL, LL, LL, DH,
         BB*HH, (long long)LL*LL, (long long)DH*LL, (long long)LL*DH, 1.0f, 0);
    k_merge<<<dim3((1 << 20) / 256), dim3(256), 0, stream>>>(Om_b, Oh_f);
    gemm(y_f, nullptr, Om_b, w_out[s], T.out_b,
         NTOK, DD, DD, DD, DD, DD, 1, 0, 0, 0, 1.0f, 0);
    k_add_ln<<<dim3(NTOK), dim3(256), 0, stream>>>(x1_f, x1_b, y_f, xemb_f[s], T.g1, T.b1);
    // FFN
    gemm(hf_f, hf_b, x1_b, w_f1[s], T.bf1,
         NTOK, 2048, DD, DD, DD, 2048, 1, 0, 0, 0, 1.0f, 1);
    gemm(f_f, nullptr, hf_b, w_f2[s], T.bf2,
         NTOK, DD, 2048, 2048, 2048, DD, 1, 0, 0, 0, 1.0f, 0);
    k_add_ln<<<dim3(NTOK), dim3(256), 0, stream>>>(x2_f, nullptr, f_f, x1_f, T.g2, T.b2);
    k_add_ln<<<dim3(NTOK), dim3(256), 0, stream>>>(enc_f, nullptr, x2_f, nullptr, T.gn, T.bn);
    // res = [enc | emb]  (+ ctx^T for d/sd, + rowsums)
    k_concat<<<dim3(NTOK), dim3(256), 0, stream>>>(
        resf[s], resb[s], (s == 0) ? nullptr : ctxT[s - 1], rowsum[s], enc_f, xemb_f[s]);
  }

  // ---- BiDAF branches (ctx = d_res, sd_res) ----
  for (int br = 0; br < 2; ++br) {
    const int s = br + 1;
    const u16*   Wb = br ? w_qsd : w_qd;
    const float* Bb = br ? qsd_b : qd_b;
    // S = ctx @ q_res^T   (batched over B)
    gemm(Sb_f, nullptr, resb[s], resb[0], nullptr, LL, LL, 512, 512, 512, LL,
         BB, (long long)LL*512, (long long)LL*512, (long long)LL*LL, 1.0f, 0);
    k_sepi<<<dim3(BB*LL), dim3(256), 0, stream>>>(Sb_f, rowsum[s], rowsum[0]);
    k_argmax<<<dim3(BB*LL), dim3(256), 0, stream>>>(idxf, Sb_f);
    k_softmax<<<dim3(BB*LL), dim3(256), 0, stream>>>(nullptr, Pb_b, Sb_f);
    // c2q = softmax(S) @ ctx   (W = ctx^T stored [512, L])
    gemm(c2q_f, nullptr, Pb_b, ctxT[br], nullptr, LL, 512, LL, LL, LL, 512,
         BB, (long long)LL*LL, (long long)512*LL, (long long)LL*512, 1.0f, 0);
    k_softmax<<<dim3(BB), dim3(256), 0, stream>>>(beta, nullptr, idxf);  // over T
    k_q2c<<<dim3(BB * 512 / 256), dim3(256), 0, stream>>>(q2c, beta, resf[0]);
    k_cat<<<dim3((unsigned)(N * 3072 / 256)), dim3(256), 0, stream>>>(
        cat_b, resf[0], resf[s], c2q_f, q2c);
    // branch out -> proper half of allc (bf16 only; feeds next GEMM)
    gemm(nullptr, allc_b + br * 3072, cat_b, Wb, Bb,
         NTOK, 3072, 3072, 3072, 3072, 6144, 1, 0, 0, 0, 1.0f, 0);
  }

  // ---- residual projection + pooling + classifier ----
  gemm(rout_f, nullptr, allc_b, w_res, p_res_b,
       NTOK, 6144, 6144, 6144, 6144, 6144, 1, 0, 0, 0, 1.0f, 0);
  k_pool<<<dim3(BB * 6144 / 256), dim3(256), 0, stream>>>(r_f, r_b, rout_f);
  gemm(nullptr, h1_b, r_b, w_c1, c_b1, BB, 8192, 12288, 12288, 12288, 8192,
       1, 0, 0, 0, 1.0f, 1);
  gemm(nullptr, h2_b, h1_b, w_c2, c_b2, BB, 2048, 8192, 8192, 8192, 2048,
       1, 0, 0, 0, 1.0f, 1);
  gemm(nullptr, h3_b, h2_b, w_c3, c_b3, BB, 1024, 2048, 2048, 2048, 1024,
       1, 0, 0, 0, 1.0f, 1);
  gemm((float*)d_out, nullptr, h3_b, w_c4, c_b4, BB, 5, 1024, 1024, 1024, 5,
       1, 0, 0, 0, 1.0f, 0);
}
